// SyndromeTransformerLayer_29411936043725
// MI455X (gfx1250) — matrix operand order repaired and checked by reading, not compiled, not run on hardware
//
#include <hip/hip_runtime.h>
#include <hip/hip_bf16.h>
#include <math.h>

typedef __attribute__((ext_vector_type(16))) __bf16 v16bf;
typedef __attribute__((ext_vector_type(8)))  float  v8f;

#define B_   64
#define S_   575
#define D_   256
#define H_   8
#define HD_  32
#define FF_  1024
#define P_   576          // padded sequence (24*24)
#define M_   (B_*S_)      // 36800 rows = 1150 * 32
#define MT32_ (M_/32)     // 1150 M-tiles of 32 rows
#define HEADSZ_ 18400     // S_*HD_ : flat-reshape head block size

// ---------- small helpers ----------
__device__ __forceinline__ unsigned short f2bf(float f) {
  unsigned u = __float_as_uint(f);
  u += 0x7FFFu + ((u >> 16) & 1u);          // round-to-nearest-even
  return (unsigned short)(u >> 16);
}

union FragU { uint4 u[2]; v16bf v; };

// A fragment: A[m][k] row-major (ldk elems). lane&15 selects M row (caller adds),
// K packing per ISA: lanes<16 hold K {0..7,16..23}, lanes>=16 hold K {8..15,24..31}.
__device__ __forceinline__ v16bf load_frag_a(const unsigned short* X, int ldk,
                                             long m0, int kbase, int lane) {
  int hi = lane >> 4;
  const unsigned* p = (const unsigned*)(X + m0 * (long)ldk + kbase);
  FragU r;
  r.u[0] = *(const uint4*)(p + 4 * hi);
  r.u[1] = *(const uint4*)(p + 8 + 4 * hi);
  return r.v;
}

// B fragment from W[n][k] row-major (ldk elems): lane holds column n, K packed like A ({0..7,16..23} / {8..15,24..31}).
__device__ __forceinline__ v16bf load_frag_b(const unsigned short* W, int ldk,
                                             long n0, int kbase, int lane) {
  int hi = lane >> 4;
  const unsigned* p = (const unsigned*)(W + n0 * (long)ldk + kbase);
  FragU r;
  r.u[0] = *(const uint4*)(p + 4 * hi);
  r.u[1] = *(const uint4*)(p + 8 + 4 * hi);
  return r.v;
}

__device__ __forceinline__ v8f wmma_bf(v16bf a, v16bf b, v8f c) {
  return __builtin_amdgcn_wmma_f32_16x16x32_bf16(false, a, false, b, (short)0, c,
                                                 false, false);
}

// Fused residual-add + LayerNorm over 32 rows x 256 cols sitting in LDS.
// rows[r] is the global row index (b*S+s) or -1 to skip. blockDim == 256
// (8 threads per row, 32 columns each).
__device__ void epilogue_ln32(float* ldsC, float* redS, float* redQ, float* stats,
                              const long* rows, const float* resid,
                              const float* gamma, const float* beta,
                              float* out32, unsigned short* out16, int tid) {
  int r = tid >> 3, j = tid & 7;
  long row = rows[r];
  float s = 0.f, q = 0.f;
  if (row >= 0) {
    const float* rp = resid + row * 256;
    for (int k = 0; k < 32; ++k) {
      int c = j + (k << 3);
      float x = ldsC[r * 256 + c] + rp[c];
      ldsC[r * 256 + c] = x;
      s += x; q += x * x;
    }
  }
  redS[r * 8 + j] = s; redQ[r * 8 + j] = q;
  __syncthreads();
  if (tid < 32) {
    float a = 0.f, b = 0.f;
    for (int k = 0; k < 8; ++k) { a += redS[tid * 8 + k]; b += redQ[tid * 8 + k]; }
    float mu = a * (1.f / 256.f);
    float var = b * (1.f / 256.f) - mu * mu;
    stats[tid * 2] = mu;
    stats[tid * 2 + 1] = rsqrtf(var + 1e-5f);
  }
  __syncthreads();
  if (row >= 0) {
    float mu = stats[r * 2], rs = stats[r * 2 + 1];
    for (int k = 0; k < 32; ++k) {
      int c = j + (k << 3);
      float y = (ldsC[r * 256 + c] - mu) * rs * gamma[c] + beta[c];
      out32[row * 256 + c] = y;
      if (out16) out16[row * 256 + c] = f2bf(y);
    }
  }
}

// ---------- K0: weight conversion / repack to bf16 ----------
__global__ __launch_bounds__(256) void k_pack(
    const float* conv_w, const float* qkv_w, const float* o_w,
    const float* ffp_w, const float* ffg_w, const float* ffo_w,
    unsigned short* convw_r, unsigned short* qkv_bf, unsigned short* o_bf,
    unsigned short* ffp_bf, unsigned short* ffg_bf, unsigned short* ffo_bf) {
  long idx = (long)blockIdx.x * 256 + threadIdx.x;
  const long N0 = 27L * 256 * 256, N1 = 768L * 256, N2 = 256L * 256, N3 = 1024L * 256;
  if (idx < N0) {
    long tap = idx >> 16, rem = idx & 65535;
    long o = rem >> 8, ci = rem & 255;
    long i = tap / 9, kk = tap % 9;
    convw_r[idx] = f2bf(conv_w[((i * 256 + o) * 256 + ci) * 9 + kk]);
  } else if ((idx -= N0) < N1) qkv_bf[idx] = f2bf(qkv_w[idx]);
  else if ((idx -= N1) < N2)  o_bf[idx]   = f2bf(o_w[idx]);
  else if ((idx -= N2) < N3)  ffp_bf[idx] = f2bf(ffp_w[idx]);
  else if ((idx -= N3) < N3)  ffg_bf[idx] = f2bf(ffg_w[idx]);
  else if ((idx -= N3) < N3)  ffo_bf[idx] = f2bf(ffo_w[idx]);
}

// ---------- K1: 3-dilation conv as 27 shifted GEMM taps + residual + LN1 ----------
// 32-row M-tile: each wave does 2 M-frags x 2 N-tiles = 4 WMMAs per K-step,
// each B fragment (global) reused across both M fragments.
__global__ __launch_bounds__(256) void k_conv_ln1(
    const float* state, const unsigned short* convw_r, const float* conv_b,
    const float* n1_g, const float* n1_b,
    float* state1_f32, unsigned short* state1_bf16) {
  __shared__ __align__(16) unsigned short ldsA[32 * 256];   // 16 KB
  __shared__ float ldsC[32 * 256];                          // 32 KB
  __shared__ float redS[32 * 8], redQ[32 * 8], stats[64];
  __shared__ long rowsArr[32];

  int tile = blockIdx.x, b = blockIdx.y;
  int tid = threadIdx.x, lane = tid & 31, wave = tid >> 5;
  int r8 = tid >> 3, j8 = tid & 7;      // staging: row r8 (0..31), 8 threads/row
  int p = tile * 32 + r8;
  int y = p / 24, x = p % 24;

  v8f acc[2][2];
  for (int g = 0; g < 8; ++g) {
    acc[0][0][g] = 0.f; acc[0][1][g] = 0.f;
    acc[1][0][g] = 0.f; acc[1][1][g] = 0.f;
  }

  for (int i3 = 0; i3 < 3; ++i3) {
    int dil = 1 << i3;
    for (int ky = 0; ky < 3; ++ky)
      for (int kx = 0; kx < 3; ++kx) {
        int tap = i3 * 9 + ky * 3 + kx;
        int ys = y + (ky - 1) * dil, xs = x + (kx - 1) * dil;
        bool ok = ((unsigned)ys < 24u) && ((unsigned)xs < 24u);
        int p2 = ys * 24 + xs;
        bool okr = ok && (p2 >= 1);
        const float* src = state + ((long)b * S_ + (p2 - 1)) * D_;
        for (int k = 0; k < 32; ++k) {
          int ci = j8 + (k << 3);
          ldsA[r8 * 256 + ci] = okr ? f2bf(src[ci]) : (unsigned short)0;
        }
        __syncthreads();
        const unsigned short* Wt = convw_r + (long)tap * 65536;
        for (int ks = 0; ks < 8; ++ks) {
          v16bf af0 = load_frag_a(ldsA, 256, (long)(lane & 15), ks * 32, lane);
          v16bf af1 = load_frag_a(ldsA, 256, (long)(16 + (lane & 15)), ks * 32, lane);
          for (int t2 = 0; t2 < 2; ++t2) {
            long n0 = (long)(wave * 32 + t2 * 16 + (lane & 15));
            v16bf bf = load_frag_b(Wt, 256, n0, ks * 32, lane);
            acc[0][t2] = wmma_bf(af0, bf, acc[0][t2]);
            acc[1][t2] = wmma_bf(af1, bf, acc[1][t2]);
          }
        }
        __syncthreads();
      }
  }
  {
    int hi = lane >> 4, nl = lane & 15;
    for (int t2 = 0; t2 < 2; ++t2) {
      int n = wave * 32 + t2 * 16 + nl;
      float cb = conv_b[n] + conv_b[256 + n] + conv_b[512 + n];
      for (int mf = 0; mf < 2; ++mf)
        for (int g = 0; g < 8; ++g)
          ldsC[(mf * 16 + g + 8 * hi) * 256 + n] = acc[mf][t2][g] + cb;
    }
    if (tid < 32) {
      int pp = tile * 32 + tid;
      rowsArr[tid] = (pp >= 1) ? ((long)b * S_ + (pp - 1)) : -1L;
    }
  }
  __syncthreads();
  epilogue_ln32(ldsC, redS, redQ, stats, rowsArr, state, n1_g, n1_b,
                state1_f32, state1_bf16, tid);
}

// ---------- K2: QKV projection + flat-reshape head split ----------
__global__ __launch_bounds__(256) void k_qkv(
    const unsigned short* state1_bf16, const unsigned short* qkv_bf, const float* qkv_b,
    unsigned short* q_r, unsigned short* k_r, unsigned short* v_t) {
  int tile = blockIdx.x, sec = blockIdx.y;   // sec: 0=q,1=k,2=v
  int tid = threadIdx.x, lane = tid & 31, wave = tid >> 5;
  long mbase = (long)tile * 32;
  v8f acc[2][2];
  for (int g = 0; g < 8; ++g) {
    acc[0][0][g] = 0.f; acc[0][1][g] = 0.f;
    acc[1][0][g] = 0.f; acc[1][1][g] = 0.f;
  }
  for (int ks = 0; ks < 8; ++ks) {
    v16bf af0 = load_frag_a(state1_bf16, 256, mbase + (lane & 15), ks * 32, lane);
    v16bf af1 = load_frag_a(state1_bf16, 256, mbase + 16 + (lane & 15), ks * 32, lane);
    for (int t2 = 0; t2 < 2; ++t2) {
      long n0 = (long)sec * 256 + wave * 32 + t2 * 16 + (lane & 15);
      v16bf bf = load_frag_b(qkv_bf, 256, n0, ks * 32, lane);
      acc[0][t2] = wmma_bf(af0, bf, acc[0][t2]);
      acc[1][t2] = wmma_bf(af1, bf, acc[1][t2]);
    }
  }
  int hi = lane >> 4, nl = lane & 15;
  for (int t2 = 0; t2 < 2; ++t2) {
    int col = wave * 32 + t2 * 16 + nl;        // 0..255 within section
    float bias = qkv_b[sec * 256 + col];
    for (int mf = 0; mf < 2; ++mf)
      for (int g = 0; g < 8; ++g) {
        long m = mbase + mf * 16 + g + 8 * hi;
        int b = (int)(m / S_), s = (int)(m % S_);
        float val = acc[mf][t2][g] + bias;
        // reference reshape (b,s,d)->(b,h,s,hd) is a flat reinterpretation:
        long flat = (long)s * 256 + col;
        int h = (int)(flat / HEADSZ_);
        long rem = flat - (long)h * HEADSZ_;
        int s2 = (int)(rem >> 5), e = (int)(rem & 31);
        long bh = (long)(b * H_ + h);
        if (sec == 0)      q_r[(bh * S_ + s2) * HD_ + e] = f2bf(val * 0.17677669529663687f);
        else if (sec == 1) k_r[(bh * S_ + s2) * HD_ + e] = f2bf(val);
        else               v_t[(bh * HD_ + e) * P_ + s2] = f2bf(val);
      }
  }
}

// ---------- K3: flash attention, one wave per (b,h,16-query tile) ----------
__global__ __launch_bounds__(32) void k_attn(
    const unsigned short* q_r, const unsigned short* k_r, const unsigned short* v_t,
    const float* attn_bias, unsigned short* attn_out) {
  __shared__ __align__(16) unsigned short ldsP[16 * 32];
  int bid = blockIdx.x;
  int mtile = bid % 36;
  int h = (bid / 36) % H_;
  int b = bid / (36 * H_);
  int lane = threadIdx.x;
  int hi = lane >> 4, nl = lane & 15;
  int s0 = mtile * 16;
  long bh = (long)(b * H_ + h);
  const unsigned short* qp = q_r + bh * S_ * HD_;
  const unsigned short* kp = k_r + bh * S_ * HD_;
  const unsigned short* vp = v_t + bh * HD_ * P_;
  const float* bp = attn_bias + (long)h * S_ * S_;

  v16bf qf = load_frag_a(qp, HD_, (long)(s0 + nl), 0, lane);

  float Mrow[8], Lrow[8];
  v8f O0, O1;
  for (int g = 0; g < 8; ++g) { Mrow[g] = -1e30f; Lrow[g] = 0.f; O0[g] = 0.f; O1[g] = 0.f; }

  for (int kb = 0; kb < 18; ++kb) {              // 18 * 32 = 576 keys (last masked)
    v8f z; for (int g = 0; g < 8; ++g) z[g] = 0.f;
    v16bf kf0 = load_frag_b(kp, HD_, (long)(kb * 32 + nl), 0, lane);
    v16bf kf1 = load_frag_b(kp, HD_, (long)(kb * 32 + 16 + nl), 0, lane);
    v8f sc0 = wmma_bf(qf, kf0, z);
    v8f sc1 = wmma_bf(qf, kf1, z);

    float nm[8];
    int t0g = kb * 32 + nl, t1g = t0g + 16;
    for (int g = 0; g < 8; ++g) {
      int s = s0 + g + 8 * hi;
      int sb = (s < S_) ? s : (S_ - 1);
      float v0 = sc0[g], v1 = sc1[g];
      v0 += (t0g < S_) ? bp[(long)sb * S_ + t0g] : 0.f;
      v1 += (t1g < S_) ? bp[(long)sb * S_ + t1g] : 0.f;
      if (t0g >= S_) v0 = -1e30f;
      if (t1g >= S_) v1 = -1e30f;
      sc0[g] = v0; sc1[g] = v1;
      float mx = fmaxf(v0, v1);
      for (int o = 8; o > 0; o >>= 1) mx = fmaxf(mx, __shfl_xor(mx, o, 32));
      nm[g] = fmaxf(Mrow[g], mx);
    }
    for (int g = 0; g < 8; ++g) {
      float sc = __expf(Mrow[g] - nm[g]);
      float p0 = __expf(sc0[g] - nm[g]);
      float p1 = __expf(sc1[g] - nm[g]);
      float rs = p0 + p1;
      for (int o = 8; o > 0; o >>= 1) rs += __shfl_xor(rs, o, 32);
      Lrow[g] = Lrow[g] * sc + rs;
      Mrow[g] = nm[g];
      O0[g] *= sc; O1[g] *= sc;
      ldsP[(g + 8 * hi) * 32 + nl]      = f2bf(p0);
      ldsP[(g + 8 * hi) * 32 + 16 + nl] = f2bf(p1);
    }
    asm volatile("s_wait_dscnt 0" ::: "memory");
    v16bf pf  = load_frag_a(ldsP, 32, (long)nl, 0, lane);
    v16bf vf0 = load_frag_b(vp, P_, (long)nl, kb * 32, lane);
    v16bf vf1 = load_frag_b(vp, P_, (long)(16 + nl), kb * 32, lane);
    O0 = wmma_bf(pf, vf0, O0);
    O1 = wmma_bf(pf, vf1, O1);
    asm volatile("s_wait_dscnt 0" ::: "memory");
  }
  for (int g = 0; g < 8; ++g) {
    int s = s0 + g + 8 * hi;
    if (s < S_) {
      float rl = 1.f / Lrow[g];
      long row = (long)b * S_ + s;
      attn_out[row * 256 + h * 32 + nl]      = f2bf(O0[g] * rl);
      attn_out[row * 256 + h * 32 + 16 + nl] = f2bf(O1[g] * rl);
    }
  }
}

// ---------- K4: output projection + residual + LN2 ----------
__global__ __launch_bounds__(256) void k_oproj_ln2(
    const unsigned short* attn_out, const unsigned short* o_bf, const float* o_b,
    const float* state1_f32, const float* n2_g, const float* n2_b,
    float* state2_f32, unsigned short* state2_bf16) {
  __shared__ float ldsC[32 * 256];
  __shared__ float redS[32 * 8], redQ[32 * 8], stats[64];
  __shared__ long rowsArr[32];
  int tile = blockIdx.x;
  int tid = threadIdx.x, lane = tid & 31, wave = tid >> 5;
  long mbase = (long)tile * 32;
  v8f acc[2][2];
  for (int g = 0; g < 8; ++g) {
    acc[0][0][g] = 0.f; acc[0][1][g] = 0.f;
    acc[1][0][g] = 0.f; acc[1][1][g] = 0.f;
  }
  for (int ks = 0; ks < 8; ++ks) {
    v16bf af0 = load_frag_a(attn_out, 256, mbase + (lane & 15), ks * 32, lane);
    v16bf af1 = load_frag_a(attn_out, 256, mbase + 16 + (lane & 15), ks * 32, lane);
    for (int t2 = 0; t2 < 2; ++t2) {
      long n0 = (long)(wave * 32 + t2 * 16 + (lane & 15));
      v16bf bf = load_frag_b(o_bf, 256, n0, ks * 32, lane);
      acc[0][t2] = wmma_bf(af0, bf, acc[0][t2]);
      acc[1][t2] = wmma_bf(af1, bf, acc[1][t2]);
    }
  }
  int hi = lane >> 4, nl = lane & 15;
  for (int t2 = 0; t2 < 2; ++t2) {
    int n = wave * 32 + t2 * 16 + nl;
    float bias = o_b[n];
    for (int mf = 0; mf < 2; ++mf)
      for (int g = 0; g < 8; ++g)
        ldsC[(mf * 16 + g + 8 * hi) * 256 + n] = acc[mf][t2][g] + bias;
  }
  if (tid < 32) rowsArr[tid] = mbase + tid;
  __syncthreads();
  epilogue_ln32(ldsC, redS, redQ, stats, rowsArr, state1_f32, n2_g, n2_b,
                state2_f32, state2_bf16, tid);
}

// ---------- K5: FFN proj + sigmoid gate ----------
__global__ __launch_bounds__(256) void k_ffn1(
    const unsigned short* state2_bf16, const unsigned short* ffp_bf,
    const unsigned short* ffg_bf, const float* ffp_b, const float* ffg_b,
    unsigned short* ffh) {
  int tile = blockIdx.x, sec = blockIdx.y;   // sec covers FF cols [sec*256, +256)
  int tid = threadIdx.x, lane = tid & 31, wave = tid >> 5;
  long mbase = (long)tile * 32;
  v8f aP[2][2], aG[2][2];
  for (int g = 0; g < 8; ++g) {
    aP[0][0][g] = aP[0][1][g] = aP[1][0][g] = aP[1][1][g] = 0.f;
    aG[0][0][g] = aG[0][1][g] = aG[1][0][g] = aG[1][1][g] = 0.f;
  }
  for (int ks = 0; ks < 8; ++ks) {
    v16bf af0 = load_frag_a(state2_bf16, 256, mbase + (lane & 15), ks * 32, lane);
    v16bf af1 = load_frag_a(state2_bf16, 256, mbase + 16 + (lane & 15), ks * 32, lane);
    for (int t2 = 0; t2 < 2; ++t2) {
      long n0 = (long)sec * 256 + wave * 32 + t2 * 16 + (lane & 15);
      v16bf bP = load_frag_b(ffp_bf, 256, n0, ks * 32, lane);
      v16bf bG = load_frag_b(ffg_bf, 256, n0, ks * 32, lane);
      aP[0][t2] = wmma_bf(af0, bP, aP[0][t2]);
      aP[1][t2] = wmma_bf(af1, bP, aP[1][t2]);
      aG[0][t2] = wmma_bf(af0, bG, aG[0][t2]);
      aG[1][t2] = wmma_bf(af1, bG, aG[1][t2]);
    }
  }
  int hi = lane >> 4, nl = lane & 15;
  for (int t2 = 0; t2 < 2; ++t2) {
    int nf = sec * 256 + wave * 32 + t2 * 16 + nl;
    float pb = ffp_b[nf], gb = ffg_b[nf];
    for (int mf = 0; mf < 2; ++mf)
      for (int g = 0; g < 8; ++g) {
        long m = mbase + mf * 16 + g + 8 * hi;
        float pr = aP[mf][t2][g] + pb;
        float gt = aG[mf][t2][g] + gb;
        float sg = 1.f / (1.f + __expf(-gt));
        ffh[m * FF_ + nf] = f2bf(pr * sg);
      }
  }
}

// ---------- K6: FFN out projection + residual + LN3 -> d_out ----------
__global__ __launch_bounds__(256) void k_ffo_ln3(
    const unsigned short* ffh, const unsigned short* ffo_bf, const float* ffo_b,
    const float* state2_f32, const float* n3_g, const float* n3_b, float* out) {
  __shared__ float ldsC[32 * 256];
  __shared__ float redS[32 * 8], redQ[32 * 8], stats[64];
  __shared__ long rowsArr[32];
  int tile = blockIdx.x;
  int tid = threadIdx.x, lane = tid & 31, wave = tid >> 5;
  long mbase = (long)tile * 32;
  v8f acc[2][2];
  for (int g = 0; g < 8; ++g) {
    acc[0][0][g] = 0.f; acc[0][1][g] = 0.f;
    acc[1][0][g] = 0.f; acc[1][1][g] = 0.f;
  }
  for (int ks = 0; ks < 32; ++ks) {
    v16bf af0 = load_frag_a(ffh, FF_, mbase + (lane & 15), ks * 32, lane);
    v16bf af1 = load_frag_a(ffh, FF_, mbase + 16 + (lane & 15), ks * 32, lane);
    for (int t2 = 0; t2 < 2; ++t2) {
      long n0 = (long)(wave * 32 + t2 * 16 + (lane & 15));
      v16bf bf = load_frag_b(ffo_bf, FF_, n0, ks * 32, lane);
      acc[0][t2] = wmma_bf(af0, bf, acc[0][t2]);
      acc[1][t2] = wmma_bf(af1, bf, acc[1][t2]);
    }
  }
  int hi = lane >> 4, nl = lane & 15;
  for (int t2 = 0; t2 < 2; ++t2) {
    int n = wave * 32 + t2 * 16 + nl;
    float bias = ffo_b[n];
    for (int mf = 0; mf < 2; ++mf)
      for (int g = 0; g < 8; ++g)
        ldsC[(mf * 16 + g + 8 * hi) * 256 + n] = acc[mf][t2][g] + bias;
  }
  if (tid < 32) rowsArr[tid] = mbase + tid;
  __syncthreads();
  epilogue_ln32(ldsC, redS, redQ, stats, rowsArr, state2_f32, n3_g, n3_b,
                out, nullptr, tid);
}

// ---------- host launcher ----------
extern "C" void kernel_launch(void* const* d_in, const int* in_sizes, int n_in,
                              void* d_out, int out_size, void* d_ws, size_t ws_size,
                              hipStream_t stream) {
  (void)in_sizes; (void)n_in; (void)out_size; (void)ws_size;
  const float* state     = (const float*)d_in[0];
  const float* conv_w    = (const float*)d_in[1];
  const float* conv_b    = (const float*)d_in[2];
  const float* qkv_w     = (const float*)d_in[3];
  const float* qkv_b     = (const float*)d_in[4];
  const float* o_w       = (const float*)d_in[5];
  const float* o_b       = (const float*)d_in[6];
  const float* attn_bias = (const float*)d_in[7];
  const float* ffp_w     = (const float*)d_in[8];
  const float* ffp_b     = (const float*)d_in[9];
  const float* ffg_w     = (const float*)d_in[10];
  const float* ffg_b     = (const float*)d_in[11];
  const float* ffo_w     = (const float*)d_in[12];
  const float* ffo_b     = (const float*)d_in[13];
  const float* n1_g = (const float*)d_in[14];
  const float* n1_b = (const float*)d_in[15];
  const float* n2_g = (const float*)d_in[16];
  const float* n2_b = (const float*)d_in[17];
  const float* n3_g = (const float*)d_in[18];
  const float* n3_b = (const float*)d_in[19];
  float* out = (float*)d_out;

  char* ws = (char*)d_ws;
  size_t off = 0;
  auto alloc = [&](size_t bytes) -> void* {
    void* p = ws + off;
    off += ((bytes + 1023) & ~(size_t)1023) + 4096;  // slack for tile-edge overreads
    return p;
  };
  float*          state1_f32  = (float*)alloc((size_t)M_ * 256 * 4);
  float*          state2_f32  = (float*)alloc((size_t)M_ * 256 * 4);
  unsigned short* state1_bf16 = (unsigned short*)alloc((size_t)M_ * 256 * 2);
  unsigned short* state2_bf16 = (unsigned short*)alloc((size_t)M_ * 256 * 2);
  unsigned short* attn_out    = (unsigned short*)alloc((size_t)M_ * 256 * 2);
  unsigned short* q_r         = (unsigned short*)alloc((size_t)B_ * H_ * S_ * HD_ * 2);
  unsigned short* k_r         = (unsigned short*)alloc((size_t)B_ * H_ * S_ * HD_ * 2);
  unsigned short* v_t         = (unsigned short*)alloc((size_t)B_ * H_ * HD_ * P_ * 2);
  unsigned short* ffh         = (unsigned short*)alloc((size_t)M_ * FF_ * 2);
  unsigned short* convw_r     = (unsigned short*)alloc((size_t)27 * 256 * 256 * 2);
  unsigned short* qkv_bf      = (unsigned short*)alloc((size_t)768 * 256 * 2);
  unsigned short* o_bf        = (unsigned short*)alloc((size_t)256 * 256 * 2);
  unsigned short* ffp_bf      = (unsigned short*)alloc((size_t)1024 * 256 * 2);
  unsigned short* ffg_bf      = (unsigned short*)alloc((size_t)1024 * 256 * 2);
  unsigned short* ffo_bf      = (unsigned short*)alloc((size_t)256 * 1024 * 2);

  long pack_elems = 27L * 65536 + 768L * 256 + 256L * 256 + 3L * 1024 * 256;
  k_pack<<<dim3((unsigned)((pack_elems + 255) / 256)), 256, 0, stream>>>(
      conv_w, qkv_w, o_w, ffp_w, ffg_w, ffo_w,
      convw_r, qkv_bf, o_bf, ffp_bf, ffg_bf, ffo_bf);

  k_conv_ln1<<<dim3(18, B_), 256, 0, stream>>>(
      state, convw_r, conv_b, n1_g, n1_b, state1_f32, state1_bf16);

  k_qkv<<<dim3(MT32_, 3), 256, 0, stream>>>(
      state1_bf16, qkv_bf, qkv_b, q_r, k_r, v_t);

  k_attn<<<dim3(36 * H_ * B_), 32, 0, stream>>>(
      q_r, k_r, v_t, attn_bias, attn_out);

  k_oproj_ln2<<<dim3(MT32_), 256, 0, stream>>>(
      attn_out, o_bf, o_b, state1_f32, n2_g, n2_b, state2_f32, state2_bf16);

  k_ffn1<<<dim3(MT32_, 4), 256, 0, stream>>>(
      state2_bf16, ffp_bf, ffg_bf, ffp_b, ffg_b, ffh);

  k_ffo_ln3<<<dim3(MT32_), 256, 0, stream>>>(
      ffh, ffo_bf, ffo_b, state2_f32, n3_g, n3_b, out);
}
